// MOLELinear_49529562858339
// MI455X (gfx1250) — matrix operand order repaired and checked by reading, not compiled, not run on hardware
//
#include <hip/hip_runtime.h>

// ---------------------------------------------------------------------------
// MOLE linear for MI455X (gfx1250, wave32, WMMA)
//   ws layout: [0, 128MB)        mixed weights, bf16, [G][O][I]
//              [128MB, 256MB)    x converted to bf16, [G*R][I]
//              [256MB, +256KB)   mixed bias, f32, [G][O]
// GEMM: block = 8 waves, tile 128x256, BK=32; wave tile 64x64 = 4x4 WMMA
// accumulators -> 16 v_wmma_f32_16x16x32_bf16 per wave per K-step.
// Inner loop is pure vmem/ds/wmma: all f32->bf16 conversion hoisted out.
// ---------------------------------------------------------------------------

#define G_NUM  64
#define R_PER  1024
#define IN_F   1024
#define OUT_F  1024
#define N_EXP  8

#define BM 128
#define BN 256
#define BK 32
#define LDS_STRIDE 40   // ushorts per row: 32 data + 8 pad (80B, 16B-aligned, bank-staggered)

typedef __attribute__((ext_vector_type(16))) __bf16         v16bf;
typedef __attribute__((ext_vector_type(8)))  float          v8f;
typedef __attribute__((ext_vector_type(8)))  unsigned short usv8;

// two f32 -> packed 2xbf16 dword (round-half-up) via v_perm_b32: 3 VALU ops
__device__ __forceinline__ unsigned int pkbf(float f0, float f1) {
    unsigned int u0 = __builtin_bit_cast(unsigned int, f0) + 0x8000u;
    unsigned int u1 = __builtin_bit_cast(unsigned int, f1) + 0x8000u;
    return __builtin_amdgcn_perm(u1, u0, 0x07060302u);  // {u1.b3,u1.b2,u0.b3,u0.b2}
}

// ---------------------------------------------------------------------------
// Kernel 0: x (f32) -> x_bf16, 8 elements/thread
// ---------------------------------------------------------------------------
__global__ __launch_bounds__(256) void mole_cvt_x(
    const float* __restrict__ x, unsigned short* __restrict__ xb)
{
    const size_t idx = ((size_t)blockIdx.x * 256 + threadIdx.x) * 8;
    const float4* p = (const float4*)(x + idx);
    float4 a = p[0], b = p[1];
    uint4 u;
    u.x = pkbf(a.x, a.y); u.y = pkbf(a.z, a.w);
    u.z = pkbf(b.x, b.y); u.w = pkbf(b.z, b.w);
    *(uint4*)(xb + idx) = u;
}

// ---------------------------------------------------------------------------
// Kernel 1: mixed_w[g][o][i] = sum_e c[g,e] * W_e[o,i] + W_sh[o,i]  (bf16 out)
// Each thread owns 8 consecutive i for one o; reads experts once, writes 64 g.
// ---------------------------------------------------------------------------
__global__ __launch_bounds__(256) void mole_mix_w(
    const float* __restrict__ coeff,            // [G,E]
    const float* __restrict__ wexp,             // [E,O,I]
    const float* __restrict__ wsh,              // [1,O,I]
    unsigned short* __restrict__ mw)            // [G,O,I] bf16
{
    __shared__ float cs[G_NUM * N_EXP];
    const int tid = threadIdx.x;
    for (int i = tid; i < G_NUM * N_EXP; i += 256) cs[i] = coeff[i];
    __syncthreads();

    const long long idx = (long long)blockIdx.x * 256 + tid;   // O*I/8 threads
    const int o  = (int)(idx >> 7);
    const int i0 = ((int)idx & 127) * 8;
    const size_t base = (size_t)o * IN_F + i0;

    float we[N_EXP][8];
    float ws[8];
#pragma unroll
    for (int e = 0; e < N_EXP; ++e) {
        const float4* p = (const float4*)(wexp + (size_t)e * OUT_F * IN_F + base);
        float4 a = p[0], b = p[1];
        we[e][0] = a.x; we[e][1] = a.y; we[e][2] = a.z; we[e][3] = a.w;
        we[e][4] = b.x; we[e][5] = b.y; we[e][6] = b.z; we[e][7] = b.w;
    }
    {
        const float4* p = (const float4*)(wsh + base);
        float4 a = p[0], b = p[1];
        ws[0] = a.x; ws[1] = a.y; ws[2] = a.z; ws[3] = a.w;
        ws[4] = b.x; ws[5] = b.y; ws[6] = b.z; ws[7] = b.w;
    }

    for (int g = 0; g < G_NUM; ++g) {
        float acc[8];
#pragma unroll
        for (int j = 0; j < 8; ++j) acc[j] = ws[j];
#pragma unroll
        for (int e = 0; e < N_EXP; ++e) {
            const float ce = cs[g * N_EXP + e];
#pragma unroll
            for (int j = 0; j < 8; ++j) acc[j] += ce * we[e][j];
        }
        uint4 ov;
        ov.x = pkbf(acc[0], acc[1]);
        ov.y = pkbf(acc[2], acc[3]);
        ov.z = pkbf(acc[4], acc[5]);
        ov.w = pkbf(acc[6], acc[7]);
        *(uint4*)(mw + (size_t)g * OUT_F * IN_F + base) = ov;
    }
}

// ---------------------------------------------------------------------------
// Kernel 2: mixed_b[g][o] = sum_e c[g,e] * b_e[o] + b_sh[o]  (f32)
// ---------------------------------------------------------------------------
__global__ __launch_bounds__(256) void mole_mix_b(
    const float* __restrict__ coeff,
    const float* __restrict__ bexp,             // [E,O]
    const float* __restrict__ bsh,              // [1,O]
    float* __restrict__ mb)                     // [G,O]
{
    const int idx = blockIdx.x * 256 + threadIdx.x;   // G*O threads
    const int g = idx >> 10;
    const int o = idx & (OUT_F - 1);
    float acc = bsh[o];
#pragma unroll
    for (int e = 0; e < N_EXP; ++e) acc += coeff[g * N_EXP + e] * bexp[e * OUT_F + o];
    mb[idx] = acc;
}

// ---------------------------------------------------------------------------
// Kernel 3: per-graph GEMM  out[g] = x_g @ W_g^T + b_g  via bf16 WMMA.
// ---------------------------------------------------------------------------
__global__ __launch_bounds__(256) void mole_gemm(
    const unsigned short* __restrict__ xb,      // [G*R, I]  bf16
    const unsigned short* __restrict__ mw,      // [G,O,I]   bf16
    const float* __restrict__ mb,               // [G,O]
    float* __restrict__ out)                    // [G*R, O]
{
    __shared__ unsigned short ldsA[BM * LDS_STRIDE];   // x tile   [r][k]
    __shared__ unsigned short ldsB[BN * LDS_STRIDE];   // W tile   [o][k]

    const int tid  = threadIdx.x;
    const int lane = tid & 31;
    const int wave = tid >> 5;
    const int wm   = wave >> 2;                 // 0..1 -> 64-row slab
    const int wn   = wave & 3;                  // 0..3 -> 64-col slab
    const int g    = blockIdx.z;
    const int m0   = blockIdx.y * BM;
    const int n0   = blockIdx.x * BN;

    // loader mapping: A: 2 threads/row of 128 rows (32B each); B: 1 thread/row of 256 (64B)
    const int arow  = tid >> 1;                 // 0..127
    const int ahalf = (tid & 1) * 16;           // 0 or 16
    const unsigned short* xrow = xb + ((size_t)g * R_PER + m0 + arow) * IN_F + ahalf;
    const unsigned short* brow = mw + ((size_t)g * OUT_F + n0 + tid) * IN_F;

    const int hs     = (lane < 16) ? 0 : 1;     // lane half-select
    const int lane16 = lane & 15;
    const int mbase  = wm * 64;
    const int nbase  = wn * 64;

    // ---- accumulators initialized with mixed bias (same o for all 8 elems) ----
    v8f acc[4][4];
#pragma unroll
    for (int ns = 0; ns < 4; ++ns) {
        const float bv = mb[g * OUT_F + n0 + nbase + ns * 16 + lane16];
        const v8f bsplat = (v8f)(bv);
#pragma unroll
        for (int ms = 0; ms < 4; ++ms) acc[ms][ns] = bsplat;
    }

    union FragU { v16bf bf; usv8 h[2]; };

    // ---- prologue: stage tile 0 into registers ----
    uint4 sa0, sa1, sb0, sb1, sb2, sb3;
    {
        const uint4* ap = (const uint4*)(xrow);
        sa0 = ap[0]; sa1 = ap[1];
        const uint4* bp = (const uint4*)(brow);
        sb0 = bp[0]; sb1 = bp[1]; sb2 = bp[2]; sb3 = bp[3];
    }

    for (int kk = 0; kk < IN_F; kk += BK) {
        // ---- staged regs -> LDS ----
        *(uint4*)&ldsA[arow * LDS_STRIDE + ahalf]     = sa0;
        *(uint4*)&ldsA[arow * LDS_STRIDE + ahalf + 8] = sa1;
        *(uint4*)&ldsB[tid * LDS_STRIDE + 0]  = sb0;
        *(uint4*)&ldsB[tid * LDS_STRIDE + 8]  = sb1;
        *(uint4*)&ldsB[tid * LDS_STRIDE + 16] = sb2;
        *(uint4*)&ldsB[tid * LDS_STRIDE + 24] = sb3;
        __syncthreads();

        // ---- stage tile k+1 (loads overlap the MMA block below) ----
        if (kk + BK < IN_F) {
            const uint4* ap = (const uint4*)(xrow + kk + BK);
            sa0 = ap[0]; sa1 = ap[1];
            const uint4* bp = (const uint4*)(brow + kk + BK);
            sb0 = bp[0]; sb1 = bp[1]; sb2 = bp[2]; sb3 = bp[3];
            if (kk + 2 * BK < IN_F) {
                __builtin_prefetch(xrow + kk + 2 * BK, 0, 1);
                __builtin_prefetch(brow + kk + 2 * BK, 0, 1);
            }
        }

        // ---- LDS -> fragments (per ISA 16-bit A/B lane layouts) ----
        FragU afr[4], bfr[4];
#pragma unroll
        for (int ms = 0; ms < 4; ++ms) {
            const int row = mbase + ms * 16 + lane16;     // M = lane%16
            const int kA  = hs * 8;                       // K groups 0-7/16-23 vs 8-15/24-31
            afr[ms].h[0] = *(const usv8*)&ldsA[row * LDS_STRIDE + kA];
            afr[ms].h[1] = *(const usv8*)&ldsA[row * LDS_STRIDE + kA + 16];
        }
#pragma unroll
        for (int ns = 0; ns < 4; ++ns) {
            const int row = nbase + ns * 16 + lane16;     // N = lane%16
            const int kB  = hs * 8;                       // K groups 0-7/16-23 vs 8-15/24-31
            bfr[ns].h[0] = *(const usv8*)&ldsB[row * LDS_STRIDE + kB];
            bfr[ns].h[1] = *(const usv8*)&ldsB[row * LDS_STRIDE + kB + 16];
        }

        // ---- 16 WMMAs per wave per K-step ----
#pragma unroll
        for (int ms = 0; ms < 4; ++ms)
#pragma unroll
            for (int ns = 0; ns < 4; ++ns)
                acc[ms][ns] = __builtin_amdgcn_wmma_f32_16x16x32_bf16(
                    false, afr[ms].bf, false, bfr[ns].bf,
                    (short)0, acc[ms][ns], false, false);

        __syncthreads();
    }

    // ---- epilogue: store f32 (D layout: M = v + hs*8, N = lane%16); bias already in ----
#pragma unroll
    for (int ns = 0; ns < 4; ++ns) {
        const int o = n0 + nbase + ns * 16 + lane16;
#pragma unroll
        for (int ms = 0; ms < 4; ++ms) {
#pragma unroll
            for (int v = 0; v < 8; ++v) {
                const int row = m0 + mbase + ms * 16 + v + hs * 8;
                out[((size_t)g * R_PER + row) * OUT_F + o] = acc[ms][ns][v];
            }
        }
    }
}

// ---------------------------------------------------------------------------
extern "C" void kernel_launch(void* const* d_in, const int* in_sizes, int n_in,
                              void* d_out, int out_size, void* d_ws, size_t ws_size,
                              hipStream_t stream) {
    (void)in_sizes; (void)n_in; (void)out_size; (void)ws_size;

    const float* x     = (const float*)d_in[0];
    const float* coeff = (const float*)d_in[1];
    const float* wexp  = (const float*)d_in[2];
    const float* bexp  = (const float*)d_in[3];
    const float* wsh   = (const float*)d_in[4];
    const float* bsh   = (const float*)d_in[5];
    float* out = (float*)d_out;

    const size_t mwBytes = (size_t)G_NUM * OUT_F * IN_F * sizeof(unsigned short); // 128MB
    const size_t xbBytes = (size_t)G_NUM * R_PER * IN_F * sizeof(unsigned short); // 128MB
    unsigned short* mw = (unsigned short*)d_ws;
    unsigned short* xbuf = (unsigned short*)((char*)d_ws + mwBytes);
    float* mbias = (float*)((char*)d_ws + mwBytes + xbBytes);                     // 256KB

    // 0) x -> bf16
    mole_cvt_x<<<(size_t)G_NUM * R_PER * IN_F / 8 / 256, 256, 0, stream>>>(x, xbuf);
    // 1) mix weights -> bf16 [G,O,I]
    mole_mix_w<<<(OUT_F * IN_F / 8) / 256, 256, 0, stream>>>(coeff, wexp, wsh, mw);
    // 2) mix biases -> f32 [G,O]
    mole_mix_b<<<(G_NUM * OUT_F) / 256, 256, 0, stream>>>(coeff, bexp, bsh, mbias);
    // 3) batched GEMM via WMMA (128x256 block tile, 64x64 wave tile)
    dim3 grid(OUT_F / BN, R_PER / BM, G_NUM);
    mole_gemm<<<grid, 256, 0, stream>>>(xbuf, mw, mbias, out);
}
